// PointNet2_54133767799199
// MI455X (gfx1250) — matrix operand order repaired and checked by reading, not compiled, not run on hardware
//
#include <hip/hip_runtime.h>
#include <hip/hip_bf16.h>
#include <stdint.h>

typedef _Float16 v16h __attribute__((ext_vector_type(16)));
typedef _Float16 v8h  __attribute__((ext_vector_type(8)));
typedef _Float16 v4h  __attribute__((ext_vector_type(4)));
typedef float    v8f  __attribute__((ext_vector_type(8)));
typedef int      v4i  __attribute__((ext_vector_type(4)));

#define AS1 __attribute__((address_space(1)))
#define AS3 __attribute__((address_space(3)))

#if __has_builtin(__builtin_amdgcn_global_load_async_to_lds_b128)
#define USE_ASYNC_LDS 1
#else
#define USE_ASYNC_LDS 0
#endif

__device__ __forceinline__ void wait_async_lds() {
#if __has_builtin(__builtin_amdgcn_s_wait_asynccnt)
    __builtin_amdgcn_s_wait_asynccnt(0);
#else
    asm volatile("s_wait_asynccnt 0" ::: "memory");
#endif
}

#define EPSV   1e-5f
#define SLOPE  0.1f

#define BATCH 4
#define NPTS1 8192
#define NCEN1 2048
#define NSAMP 32
#define NPTS2 2048   // == NCEN1
#define NCEN2 512
#define MROWS1 (BATCH * NCEN1 * NSAMP)   // 262144
#define MROWS2 (BATCH * NCEN2 * NSAMP)   // 65536

#define KMAX   160
#define LDSPAD 8

// ---------------------------------------------------------------------------
// Utility: zero a float buffer (accumulators must be re-zeroed every call)
// ---------------------------------------------------------------------------
__global__ void zero_f32_kernel(float* __restrict__ p, int n) {
    int t = blockIdx.x * blockDim.x + threadIdx.x;
    if (t < n) p[t] = 0.0f;
}

// ---------------------------------------------------------------------------
// Furthest point sampling: one workgroup per batch, whole cloud LDS-resident
// (x,y,z,dist = 16B/point -> 128KB for N=8192, fits 320KB WGP LDS).
// Sequential critical path: each of npoint steps touches only LDS + two
// log2 wave32-shuffle reductions. Matches jnp.argmax first-max tie-breaking.
// ---------------------------------------------------------------------------
__global__ __launch_bounds__(1024)
void fps_kernel(const float* __restrict__ xyz, int N, int npoint,
                int* __restrict__ outIdx, float* __restrict__ outXyz) {
    extern __shared__ float sm[];
    float* sx = sm;
    float* sy = sx + N;
    float* sz = sy + N;
    float* sd = sz + N;
    float* rv = sd + N;          // 32 per-wave best values
    __shared__ int ri[32];
    __shared__ int sFar;

    const int b   = blockIdx.x;
    const int tid = threadIdx.x;
    const int nt  = blockDim.x;
    const float* base = xyz + (size_t)b * N * 3;

    for (int j = tid; j < N; j += nt) {
        sx[j] = base[j * 3 + 0];
        sy[j] = base[j * 3 + 1];
        sz[j] = base[j * 3 + 2];
        sd[j] = 1e10f;
    }
    if (tid == 0) sFar = 0;
    __syncthreads();

    const int lane = tid & 31;
    const int wid  = tid >> 5;

    for (int i = 0; i < npoint; ++i) {
        const int far = sFar;
        if (tid == 0) {
            outIdx[b * npoint + i] = far;
            float* oc = outXyz + ((size_t)b * npoint + i) * 3;
            oc[0] = sx[far]; oc[1] = sy[far]; oc[2] = sz[far];
        }
        const float cx = sx[far], cy = sy[far], cz = sz[far];

        float bv = -1.0f; int bi = 0;
        for (int j = tid; j < N; j += nt) {
            float dx = sx[j] - cx, dy = sy[j] - cy, dz = sz[j] - cz;
            float d  = dx * dx + dy * dy + dz * dz;
            d = fminf(sd[j], d);
            sd[j] = d;
            if (d > bv) { bv = d; bi = j; }
        }
        for (int off = 16; off > 0; off >>= 1) {
            float ov = __shfl_down(bv, off, 32);
            int   oi = __shfl_down(bi, off, 32);
            if (ov > bv || (ov == bv && oi < bi)) { bv = ov; bi = oi; }
        }
        if (lane == 0) { rv[wid] = bv; ri[wid] = bi; }
        __syncthreads();
        if (wid == 0) {
            const int nw = nt >> 5;
            float v = (lane < nw) ? rv[lane] : -1.0f;
            int  ii = (lane < nw) ? ri[lane] : 0;
            for (int off = 16; off > 0; off >>= 1) {
                float ov = __shfl_down(v, off, 32);
                int   oi = __shfl_down(ii, off, 32);
                if (ov > v || (ov == v && oi < ii)) { v = ov; ii = oi; }
            }
            if (lane == 0) sFar = ii;
        }
        __syncthreads();
    }
}

// ---------------------------------------------------------------------------
// Ball query: one wave32 per center; ballot+popcount ordered compaction of
// the first NSAMP in-radius indices; pad with first hit (else 0).
// ---------------------------------------------------------------------------
__global__ __launch_bounds__(256)
void ball_query_kernel(const float* __restrict__ xyz,
                       const float* __restrict__ newXyz,
                       int N, int P, int BP, float r2,
                       int* __restrict__ outIdx) {
    const int gw   = (blockIdx.x * blockDim.x + threadIdx.x) >> 5;
    const int lane = threadIdx.x & 31;
    if (gw >= BP) return;
    const int b = gw / P;
    const float* q  = newXyz + (size_t)gw * 3;
    const float qx = q[0], qy = q[1], qz = q[2];
    const float* xb = xyz + (size_t)b * N * 3;
    int* out = outIdx + (size_t)gw * NSAMP;

    int count = 0;
    int fill = 0;
    bool haveFill = false;
    for (int base = 0; base < N && count < NSAMP; base += 32) {
        const int j = base + lane;
        bool hit = false;
        if (j < N) {
            float dx = xb[j * 3 + 0] - qx;
            float dy = xb[j * 3 + 1] - qy;
            float dz = xb[j * 3 + 2] - qz;
            hit = (dx * dx + dy * dy + dz * dz) < r2;
        }
        unsigned long long m = __ballot(hit ? 1 : 0);
        if (hit) {
            int pos = count + __popcll(m & ((1ull << lane) - 1ull));
            if (pos < NSAMP) out[pos] = j;
        }
        if (!haveFill && m != 0ull) {
            haveFill = true;
            fill = base + (__ffsll((unsigned long long)m) - 1);
        }
        count += __popcll(m);
    }
    if (count < NSAMP) {
        const int f = haveFill ? fill : 0;
        for (int k = count + lane; k < NSAMP; k += 32) out[k] = f;
    }
}

// ---------------------------------------------------------------------------
// Gather stage 1: rows = [rel(3) | xyz(3) | zero-pad to 32] as f16.
// ---------------------------------------------------------------------------
__global__ __launch_bounds__(256)
void gather1_kernel(const float* __restrict__ xyz,
                    const float* __restrict__ newXyz,
                    const int* __restrict__ idx,
                    _Float16* __restrict__ Xh) {
    const int t = blockIdx.x * blockDim.x + threadIdx.x;
    if (t >= MROWS1) return;
    const int bp = t >> 5;
    const int b  = bp / NCEN1;
    const int j  = idx[t];
    const float* pt = xyz + ((size_t)b * NPTS1 + j) * 3;
    const float* c  = newXyz + (size_t)bp * 3;
    _Float16* row = Xh + (size_t)t * 32;
    const float px = pt[0], py = pt[1], pz = pt[2];
    row[0] = (_Float16)(px - c[0]);
    row[1] = (_Float16)(py - c[1]);
    row[2] = (_Float16)(pz - c[2]);
    row[3] = (_Float16)px;
    row[4] = (_Float16)py;
    row[5] = (_Float16)pz;
    #pragma unroll
    for (int k = 6; k < 32; ++k) row[k] = (_Float16)0.0f;
}

// ---------------------------------------------------------------------------
// Gather stage 2: rows = [rel(3) | f1(128) | zero-pad to 160] as f16.
// ---------------------------------------------------------------------------
__global__ __launch_bounds__(256)
void gather2_kernel(const float* __restrict__ xyz1,
                    const float* __restrict__ newXyz2,
                    const int* __restrict__ idx,
                    const float* __restrict__ f1,
                    _Float16* __restrict__ Xh) {
    const int t = blockIdx.x * blockDim.x + threadIdx.x;
    if (t >= MROWS2) return;
    const int bp = t >> 5;
    const int b  = bp / NCEN2;
    const int j  = idx[t];
    const float* pt = xyz1 + ((size_t)b * NPTS2 + j) * 3;
    const float* c  = newXyz2 + (size_t)bp * 3;
    const float* fr = f1 + ((size_t)b * NPTS2 + j) * 128;
    _Float16* row = Xh + (size_t)t * 160;
    row[0] = (_Float16)(pt[0] - c[0]);
    row[1] = (_Float16)(pt[1] - c[1]);
    row[2] = (_Float16)(pt[2] - c[2]);
    for (int k = 0; k < 128; ++k) row[3 + k] = (_Float16)fr[k];
    #pragma unroll
    for (int k = 131; k < 160; ++k) row[k] = (_Float16)0.0f;
}

// ---------------------------------------------------------------------------
// WMMA GEMM:  Y[M,O] = Xh[M,Kpad](f16) x W[O,K](f32->f16)^T, f32 accumulate,
// + per-channel sum / sum-of-squares for BatchNorm.
// Block = 256 thr = 8 waves -> 64(M) x 64(O) tile. Each wave owns two 16x16
// D tiles sharing one B fragment (3 ds_load_b128 per wmma). A tile is a pure
// f16 copy -> staged with GLOBAL_LOAD_ASYNC_TO_LDS_B128 (ASYNCcnt) when the
// toolchain exposes it; B needs f32->f16 conversion so it stays on VALU.
// Fragment builds follow ISA 7.12.2 lane striping exactly.
// ---------------------------------------------------------------------------
__global__ __launch_bounds__(256)
void wmma_mlp_gemm_kernel(const _Float16* __restrict__ Xh,
                          const float* __restrict__ W,
                          float* __restrict__ Y,
                          float* __restrict__ sumO,
                          float* __restrict__ sumsqO,
                          int M, int K, int Kpad, int O) {
    __shared__ _Float16 Ah[64 * (KMAX + LDSPAD)];
    __shared__ _Float16 Bh[64 * (KMAX + LDSPAD)];
    __shared__ float csum[64];
    __shared__ float csq[64];

    const int tid = threadIdx.x;
    const int m0  = blockIdx.x * 64;
    const int o0  = blockIdx.y * 64;
    const int lda = Kpad + LDSPAD;

    if (tid < 64) { csum[tid] = 0.0f; csq[tid] = 0.0f; }

    // Stage A tile: 64 rows x Kpad halves, 16B chunks (nA8 multiple of 256)
    const int nA8 = (64 * Kpad) >> 3;
    for (int t8 = tid; t8 < nA8; t8 += 256) {
        const int e = t8 << 3;
        const int m = e / Kpad;
        const int k = e % Kpad;
#if USE_ASYNC_LDS
        _Float16* gp = const_cast<_Float16*>(&Xh[(size_t)(m0 + m) * Kpad + k]);
        __builtin_amdgcn_global_load_async_to_lds_b128(
            (AS1 v4i*)gp, (AS3 v4i*)&Ah[m * lda + k], 0, 0);
#else
        *(v8h*)&Ah[m * lda + k] = *(const v8h*)&Xh[(size_t)(m0 + m) * Kpad + k];
#endif
    }
    // Stage B tile: 64 out-channels x Kpad, f32 -> f16, zero K-padding
    for (int t = tid; t < 64 * Kpad; t += 256) {
        const int oc = t / Kpad;
        const int k  = t % Kpad;
        Bh[oc * lda + k] = (k < K) ? (_Float16)W[(size_t)(o0 + oc) * K + k]
                                   : (_Float16)0.0f;
    }
#if USE_ASYNC_LDS
    wait_async_lds();
#endif
    __syncthreads();

    const int wave = tid >> 5;
    const int lane = tid & 31;
    const int wm2 = wave & 1;     // which 32-row M half
    const int wo  = wave >> 1;    // O sub-tile 0..3
    const int lg  = lane >> 4;
    const int ln  = lane & 15;

    v8f acc0 = {};
    v8f acc1 = {};
    const _Float16* arow0 = &Ah[(wm2 * 32 + ln) * lda + lg * 8];
    const _Float16* arow1 = arow0 + 16 * lda;
    const _Float16* brow  = &Bh[(wo * 16 + ln) * lda + lg * 8];

    for (int kk = 0; kk < Kpad; kk += 32) {
        // B fragment: lane holds col N=ln, K = kk + lg*8 + {0..7, 16..23}
        v8h b0 = *(const v8h*)(brow + kk);
        v8h b1 = *(const v8h*)(brow + kk + 16);
        v16h bm = __builtin_shufflevector(b0, b1,
                  0,1,2,3,4,5,6,7,8,9,10,11,12,13,14,15);
        // A fragments: lane holds row M=ln, K = kk + lg*8 + {0..7, 16..23}
        v8h a0l = *(const v8h*)(arow0 + kk);
        v8h a0h = *(const v8h*)(arow0 + kk + 16);
        v16h a0 = __builtin_shufflevector(a0l, a0h,
                  0,1,2,3,4,5,6,7,8,9,10,11,12,13,14,15);
        acc0 = __builtin_amdgcn_wmma_f32_16x16x32_f16(
                   false, a0, false, bm, (short)0, acc0, false, false);
        v8h a1l = *(const v8h*)(arow1 + kk);
        v8h a1h = *(const v8h*)(arow1 + kk + 16);
        v16h a1 = __builtin_shufflevector(a1l, a1h,
                  0,1,2,3,4,5,6,7,8,9,10,11,12,13,14,15);
        acc1 = __builtin_amdgcn_wmma_f32_16x16x32_f16(
                   false, a1, false, bm, (short)0, acc1, false, false);
    }

    // D layout: VGPR r, lane -> M = r + lg*8, N = ln
    const int oc = o0 + wo * 16 + ln;
    const int mA = m0 + wm2 * 32 + lg * 8;
    float s = 0.0f, q = 0.0f;
    #pragma unroll
    for (int r = 0; r < 8; ++r) {
        const float v = acc0[r];
        Y[(size_t)(mA + r) * O + oc] = v;
        s += v; q += v * v;
    }
    #pragma unroll
    for (int r = 0; r < 8; ++r) {
        const float v = acc1[r];
        Y[(size_t)(mA + 16 + r) * O + oc] = v;
        s += v; q += v * v;
    }
    atomicAdd(&csum[wo * 16 + ln], s);   // ds_add_f32
    atomicAdd(&csq [wo * 16 + ln], q);
    __syncthreads();
    if (tid < 64) {
        atomicAdd(&sumO  [o0 + tid], csum[tid]);  // global_atomic_add_f32
        atomicAdd(&sumsqO[o0 + tid], csq[tid]);
    }
}

// ---------------------------------------------------------------------------
// BN affine + leaky ReLU fused into the f32 -> f16 conversion for the next
// layer's input. 4-wide: O is a multiple of 4 so a quad never crosses rows.
// ---------------------------------------------------------------------------
__global__ __launch_bounds__(256)
void bn_apply_f16_kernel(const float* __restrict__ Y,
                         _Float16* __restrict__ Xh,
                         const float* __restrict__ sumO,
                         const float* __restrict__ sumsqO,
                         const float* __restrict__ gam,
                         const float* __restrict__ bet,
                         int M, int O) {
    const int t4 = (blockIdx.x * blockDim.x + threadIdx.x) * 4;
    if (t4 >= M * O) return;
    const int o0 = t4 % O;
    const float invM = 1.0f / (float)M;
    v4h hv;
    #pragma unroll
    for (int u = 0; u < 4; ++u) {
        const int o = o0 + u;
        const float mean = sumO[o] * invM;
        const float var  = sumsqO[o] * invM - mean * mean;
        const float sc   = rsqrtf(var + EPSV) * gam[o];
        float v = (Y[t4 + u] - mean) * sc + bet[o];
        v = (v > 0.0f) ? v : SLOPE * v;
        hv[u] = (_Float16)v;
    }
    *(v4h*)&Xh[t4] = hv;
}

// ---------------------------------------------------------------------------
// BN + leaky ReLU + max over the sample dim (S=32). Stage-1 output (f32 f1).
// ---------------------------------------------------------------------------
__global__ __launch_bounds__(256)
void bn_maxpool_kernel(const float* __restrict__ Y,
                       const float* __restrict__ sumO,
                       const float* __restrict__ sumsqO,
                       const float* __restrict__ gam,
                       const float* __restrict__ bet,
                       float* __restrict__ outF,
                       int BP, int O, int M) {
    const int t = blockIdx.x * blockDim.x + threadIdx.x;
    if (t >= BP * O) return;
    const int o  = t % O;
    const int bp = t / O;
    const float invM = 1.0f / (float)M;
    const float mean = sumO[o] * invM;
    const float var  = sumsqO[o] * invM - mean * mean;
    const float sc   = rsqrtf(var + EPSV) * gam[o];
    const float sh   = bet[o];
    const float* yr  = Y + (size_t)bp * NSAMP * O + o;
    float best = -1e30f;
    #pragma unroll 4
    for (int s = 0; s < NSAMP; ++s) {
        float v = (yr[(size_t)s * O] - mean) * sc + sh;
        v = (v > 0.0f) ? v : SLOPE * v;
        best = fmaxf(best, v);
    }
    outF[t] = best;
}

// ---------------------------------------------------------------------------
// Stage-2 final: BN + leaky + max over S, written TRANSPOSED (B, O, P) to
// d_out with coalesced stores (thread index ordered by output layout).
// ---------------------------------------------------------------------------
__global__ __launch_bounds__(256)
void bn_maxpool_transpose_kernel(const float* __restrict__ Y,
                                 const float* __restrict__ sumO,
                                 const float* __restrict__ sumsqO,
                                 const float* __restrict__ gam,
                                 const float* __restrict__ bet,
                                 float* __restrict__ outT,
                                 int P, int O, int M) {
    const int t = blockIdx.x * blockDim.x + threadIdx.x;
    if (t >= BATCH * O * P) return;
    const int p = t % P;
    const int o = (t / P) % O;
    const int b = t / (P * O);
    const float invM = 1.0f / (float)M;
    const float mean = sumO[o] * invM;
    const float var  = sumsqO[o] * invM - mean * mean;
    const float sc   = rsqrtf(var + EPSV) * gam[o];
    const float sh   = bet[o];
    const int bp = b * P + p;
    const float* yr = Y + (size_t)bp * NSAMP * O + o;
    float best = -1e30f;
    #pragma unroll 4
    for (int s = 0; s < NSAMP; ++s) {
        float v = (yr[(size_t)s * O] - mean) * sc + sh;
        v = (v > 0.0f) ? v : SLOPE * v;
        best = fmaxf(best, v);
    }
    outT[t] = best;   // t == b*O*P + o*P + p
}

// ---------------------------------------------------------------------------
// Host-side launcher
// ---------------------------------------------------------------------------
extern "C" void kernel_launch(void* const* d_in, const int* in_sizes, int n_in,
                              void* d_out, int out_size, void* d_ws, size_t ws_size,
                              hipStream_t stream) {
    (void)in_sizes; (void)n_in; (void)out_size; (void)ws_size;

    const float* pc = (const float*)d_in[0];                 // (4,8192,3)
    const float* W1[3]  = {(const float*)d_in[1],  (const float*)d_in[2],  (const float*)d_in[3]};
    const float* Gm1[3] = {(const float*)d_in[4],  (const float*)d_in[5],  (const float*)d_in[6]};
    const float* Bt1[3] = {(const float*)d_in[7],  (const float*)d_in[8],  (const float*)d_in[9]};
    const float* W2[3]  = {(const float*)d_in[10], (const float*)d_in[11], (const float*)d_in[12]};
    const float* Gm2[3] = {(const float*)d_in[13], (const float*)d_in[14], (const float*)d_in[15]};
    const float* Bt2[3] = {(const float*)d_in[16], (const float*)d_in[17], (const float*)d_in[18]};
    float* out = (float*)d_out;

    // --- workspace carve-out (256B aligned) ---
    size_t off = 0;
    auto alloc = [&](size_t bytes) -> void* {
        void* p = (char*)d_ws + off;
        off = (off + bytes + 255) & ~(size_t)255;
        return p;
    };
    int*      idx1 = (int*)      alloc((size_t)BATCH * NCEN1 * sizeof(int));
    float*    nx1  = (float*)    alloc((size_t)BATCH * NCEN1 * 3 * sizeof(float));
    int*      bq1  = (int*)      alloc((size_t)BATCH * NCEN1 * NSAMP * sizeof(int));
    _Float16* Xh1  = (_Float16*) alloc((size_t)MROWS1 * 32 * sizeof(_Float16));
    _Float16* Hh   = (_Float16*) alloc((size_t)MROWS1 * 64 * sizeof(_Float16));   // reused
    float*    Gbig = (float*)    alloc((size_t)MROWS1 * 128 * sizeof(float));     // 128MB
    float*    Gmid = (float*)    alloc((size_t)MROWS1 * 64  * sizeof(float));     // 64MB (==M2*256)
    float*    f1   = (float*)    alloc((size_t)BATCH * NCEN1 * 128 * sizeof(float));
    int*      idx2 = (int*)      alloc((size_t)BATCH * NCEN2 * sizeof(int));
    float*    nx2  = (float*)    alloc((size_t)BATCH * NCEN2 * 3 * sizeof(float));
    int*      bq2  = (int*)      alloc((size_t)BATCH * NCEN2 * NSAMP * sizeof(int));
    _Float16* Xh2  = (_Float16*) alloc((size_t)MROWS2 * 160 * sizeof(_Float16));
    float*    acc  = (float*)    alloc((size_t)1536 * sizeof(float));

    // per-layer BN accumulators: sum at [base], sumsq at [base+O]
    const int   Osz[6]  = {64, 64, 128, 128, 128, 256};
    const int   accB[6] = {0, 128, 256, 512, 768, 1024};
    float* sum_[6]; float* sq_[6];
    for (int i = 0; i < 6; ++i) { sum_[i] = acc + accB[i]; sq_[i] = acc + accB[i] + Osz[i]; }

    // 0) zero BN accumulators (deterministic per call)
    zero_f32_kernel<<<6, 256, 0, stream>>>(acc, 1536);

    // ---- Stage 1 ----
    const size_t sh1 = ((size_t)4 * NPTS1 + 32) * sizeof(float);
    fps_kernel<<<BATCH, 1024, sh1, stream>>>(pc, NPTS1, NCEN1, idx1, nx1);

    ball_query_kernel<<<(BATCH * NCEN1 + 7) / 8, 256, 0, stream>>>(
        pc, nx1, NPTS1, NCEN1, BATCH * NCEN1, 0.2f * 0.2f, bq1);

    gather1_kernel<<<(MROWS1 + 255) / 256, 256, 0, stream>>>(pc, nx1, bq1, Xh1);

    // L0: 6(pad32) -> 64
    wmma_mlp_gemm_kernel<<<dim3(MROWS1 / 64, 1), 256, 0, stream>>>(
        Xh1, W1[0], Gmid, sum_[0], sq_[0], MROWS1, 6, 32, 64);
    bn_apply_f16_kernel<<<(MROWS1 * 64 / 4 + 255) / 256, 256, 0, stream>>>(
        Gmid, Hh, sum_[0], sq_[0], Gm1[0], Bt1[0], MROWS1, 64);

    // L1: 64 -> 64
    wmma_mlp_gemm_kernel<<<dim3(MROWS1 / 64, 1), 256, 0, stream>>>(
        Hh, W1[1], Gmid, sum_[1], sq_[1], MROWS1, 64, 64, 64);
    bn_apply_f16_kernel<<<(MROWS1 * 64 / 4 + 255) / 256, 256, 0, stream>>>(
        Gmid, Hh, sum_[1], sq_[1], Gm1[1], Bt1[1], MROWS1, 64);

    // L2: 64 -> 128
    wmma_mlp_gemm_kernel<<<dim3(MROWS1 / 64, 2), 256, 0, stream>>>(
        Hh, W1[2], Gbig, sum_[2], sq_[2], MROWS1, 64, 64, 128);
    bn_maxpool_kernel<<<(BATCH * NCEN1 * 128 + 255) / 256, 256, 0, stream>>>(
        Gbig, sum_[2], sq_[2], Gm1[2], Bt1[2], f1, BATCH * NCEN1, 128, MROWS1);

    // ---- Stage 2 ----
    const size_t sh2 = ((size_t)4 * NPTS2 + 32) * sizeof(float);
    fps_kernel<<<BATCH, 1024, sh2, stream>>>(nx1, NPTS2, NCEN2, idx2, nx2);

    ball_query_kernel<<<(BATCH * NCEN2 + 7) / 8, 256, 0, stream>>>(
        nx1, nx2, NPTS2, NCEN2, BATCH * NCEN2, 0.4f * 0.4f, bq2);

    gather2_kernel<<<(MROWS2 + 255) / 256, 256, 0, stream>>>(nx1, nx2, bq2, f1, Xh2);

    // L3: 131(pad160) -> 128
    wmma_mlp_gemm_kernel<<<dim3(MROWS2 / 64, 2), 256, 0, stream>>>(
        Xh2, W2[0], Gmid, sum_[3], sq_[3], MROWS2, 131, 160, 128);
    bn_apply_f16_kernel<<<(MROWS2 * 128 / 4 + 255) / 256, 256, 0, stream>>>(
        Gmid, Hh, sum_[3], sq_[3], Gm2[0], Bt2[0], MROWS2, 128);

    // L4: 128 -> 128
    wmma_mlp_gemm_kernel<<<dim3(MROWS2 / 64, 2), 256, 0, stream>>>(
        Hh, W2[1], Gmid, sum_[4], sq_[4], MROWS2, 128, 128, 128);
    bn_apply_f16_kernel<<<(MROWS2 * 128 / 4 + 255) / 256, 256, 0, stream>>>(
        Gmid, Hh, sum_[4], sq_[4], Gm2[1], Bt2[1], MROWS2, 128);

    // L5: 128 -> 256, then BN + leaky + max over S, transposed to (B,256,512)
    wmma_mlp_gemm_kernel<<<dim3(MROWS2 / 64, 4), 256, 0, stream>>>(
        Hh, W2[2], Gmid, sum_[5], sq_[5], MROWS2, 128, 128, 256);
    bn_maxpool_transpose_kernel<<<(BATCH * 256 * NCEN2 + 255) / 256, 256, 0, stream>>>(
        Gmid, sum_[5], sq_[5], Gm2[2], Bt2[2], out, NCEN2, 256, MROWS2);
}
